// IFAREModel_cat_5068061409507
// MI455X (gfx1250) — matrix operand order repaired and checked by reading, not compiled, not run on hardware
//
#include <hip/hip_runtime.h>
#include <hip/hip_bf16.h>
#include <math.h>

// ---------------- problem constants ----------------
constexpr int kB  = 1024;
constexpr int kS  = 61;
constexpr int kH  = 768;
constexpr int kDG = 512;
constexpr int kNL = 23;

constexpr int kKCH = kH / 32;    // 24 K-chunks of 32 for bf16 WMMA
constexpr int kNDT = kDG / 16;   // 32 d-tiles of 16

// WMMA vector types (CDNA5 / gfx1250, wave32)
typedef __attribute__((ext_vector_type(16))) __bf16 v16bf;
typedef __attribute__((ext_vector_type(8)))  float  v8f;

__device__ __forceinline__ unsigned short f32_to_bf16(float f) {
  unsigned int x = __float_as_uint(f);
  unsigned int r = x + 0x7FFFu + ((x >> 16) & 1u);   // round-to-nearest-even
  return (unsigned short)(r >> 16);
}

// ---------------------------------------------------------------------------
// Kernel A: pre-swizzle fc1_w (f32 [512,768]) into bf16 in the per-lane
// B-matrix layout of V_WMMA_F32_16X16X32_BF16:
//   chunk index = ((dt*24 + c)*32 + lane), 16 halfs per chunk
//   lane < 16 : n = lane,     k = c*32 + (i&7) + (i>=8 ? 16 : 0)      (i = 0..15)
//   lane >= 16: n = lane-16,  k = c*32 + 8 + (i&7) + (i>=8 ? 16 : 0)
//   B[k][n] = fc1_w[dt*16 + n][k]
// ---------------------------------------------------------------------------
__global__ void fc1_swizzle_kernel(const float* __restrict__ fc1_w,
                                   unsigned short* __restrict__ out) {
  int idx = blockIdx.x * 256 + threadIdx.x;           // 512*768 = 393216 total
  if (idx >= kDG * kH) return;
  int i    = idx & 15;
  int l    = (idx >> 4) & 31;
  int rest = idx >> 9;                                 // dt*24 + c
  int c    = rest % kKCH;
  int dt   = rest / kKCH;
  int d    = dt * 16 + (l & 15);
  int k    = c * 32 + ((l >= 16) ? 8 : 0) + (i & 7) + ((i & 8) << 1);
  out[idx] = f32_to_bf16(fc1_w[(size_t)d * kH + k]);
}

// ---------------------------------------------------------------------------
// Kernel B: gate. g[tok] = fc2_b + sum_d fc2_w[d] * tanh(fc1_b[d] + v[tok]·fc1_w[d])
// One workgroup (8 waves) per 16-token tile. A tile (16x768, bf16) staged in
// LDS in the WMMA A layout; each wave owns 4 d-tiles; 24 K-chunk WMMAs each.
// 62464 tokens = 3904 tiles exactly.
// ---------------------------------------------------------------------------
__global__ __launch_bounds__(256) void gate_kernel(
    const float* __restrict__ v,                 // [B*S, H]
    const unsigned short* __restrict__ fc1sw,    // swizzled bf16
    const float* __restrict__ fc1_b,
    const float* __restrict__ fc2_w,             // [512]
    const float* __restrict__ fc2_b,             // [1]
    float* __restrict__ g_out)                   // [B*S]
{
  // A tile: [c][lane] chunks, 16 halfs used, padded to 24 halfs (48 B) to cut
  // LDS bank conflicts on the ds_load_b128 pairs to 2-way.
  __shared__ __align__(16) unsigned short As[kKCH * 32 * 24];   // 36,864 B
  __shared__ float red[8][16][16];                              //  8,192 B

  const int tile = blockIdx.x;
  const int tid  = threadIdx.x;
  const int lane = tid & 31;
  const int wv   = tid >> 5;

  // Stage A-tile: per chunk (c, l): lane-group g holds K {0..7,16..23}+8g
  for (int p = tid; p < kKCH * 32; p += 256) {
    int c = p >> 5, l = p & 31;
    const float* src = v + (size_t)(tile * 16 + (l & 15)) * kH
                         + c * 32 + ((l >= 16) ? 8 : 0);
    unsigned short* dst = As + p * 24;
#pragma unroll
    for (int i = 0; i < 8; ++i) {
      dst[i]     = f32_to_bf16(src[i]);       // K = base + i
      dst[i + 8] = f32_to_bf16(src[i + 16]);  // K = base + 16 + i
    }
  }
  __syncthreads();

  float acc[8];
#pragma unroll
  for (int r = 0; r < 8; ++r) acc[r] = 0.0f;

  const int n = lane & 15;
  for (int j = 0; j < 4; ++j) {
    const int dt = wv * 4 + j;
    v8f C = {};
    const unsigned short* Bp = fc1sw + ((size_t)(dt * kKCH) * 32 + lane) * 16;
    for (int c = 0; c < kKCH; ++c) {
      v16bf a = *(const v16bf*)(As + (c * 32 + lane) * 24);
      v16bf b = *(const v16bf*)(Bp + (size_t)c * 32 * 16);
      C = __builtin_amdgcn_wmma_f32_16x16x32_bf16(
              false, a, false, b, (short)0, C, false, false);
    }
    const int d  = dt * 16 + n;
    const float b1 = fc1_b[d];
    const float w2 = fc2_w[d];
#pragma unroll
    for (int r = 0; r < 8; ++r)
      acc[r] += w2 * tanhf(C[r] + b1);
  }

  // deterministic tree reduction: acc[r] lives at row m = r + 8*(lane>=16), col n
  const int grp = lane >> 4;
#pragma unroll
  for (int r = 0; r < 8; ++r)
    red[wv][r + 8 * grp][n] = acc[r];
  __syncthreads();

  {
    int m = tid >> 4, nn = tid & 15;
    float s = 0.0f;
#pragma unroll
    for (int w = 0; w < 8; ++w) s += red[w][m][nn];
    red[0][m][nn] = s;          // each thread touches a unique (m,nn)
  }
  __syncthreads();

  if (tid < 16) {
    float s = 0.0f;
#pragma unroll
    for (int nn = 0; nn < 16; ++nn) s += red[0][tid][nn];
    g_out[tile * 16 + tid] = s + fc2_b[0];
  }
}

// ---------------------------------------------------------------------------
// Kernel C: per-batch attention (only head/tail rows) + classifier.
// mean_s == 1/61 exactly (mean of a softmax row), so:
//   a[s,t] = (q[s]·v[t] * (1-g[t]) + g[t]/61) / sqrt(768)
// One 256-thread workgroup per batch; v[b] staged once in LDS.
// ---------------------------------------------------------------------------
constexpr int kVSTR = 772;   // 768 + 4 pad floats: 2-way max bank conflicts

__global__ __launch_bounds__(256) void attn_cls_kernel(
    const float* __restrict__ q,          // [B,S,H]
    const float* __restrict__ v,          // [B,S,H]
    const float* __restrict__ g,          // [B*S]
    const int*   __restrict__ input_ids,  // [B,S]
    const int*   __restrict__ head_id_p,
    const int*   __restrict__ tail_id_p,
    const float* __restrict__ cls_w,      // [23, 1536]
    const float* __restrict__ cls_b,      // [23]
    float* __restrict__ out)              // [B, 23]
{
  extern __shared__ float sm[];
  float* v_s  = sm;                       // 61*772
  float* scr  = v_s + kS * kVSTR;         // 2*64 blended scores -> weights
  float* q_s  = scr + 128;                // 2*768 head/tail query rows
  float* ent  = q_s + 2 * kH;             // 1536 entity vector
  float* redc = ent + 2 * kH;             // 23*8 label partials
  __shared__ int hi_s, ti_s;

  const int b   = blockIdx.x;
  const int tid = threadIdx.x;
  const float inv_scale = 0.03608439182435161f;   // 1/sqrt(768)
  const float inv_s     = 1.0f / 61.0f;

  if (tid == 0) { hi_s = 0; ti_s = 0; }
  __syncthreads();
  if (tid < kS) {
    int id = input_ids[b * kS + tid];
    if (id == head_id_p[0]) hi_s = tid;   // marker is unique per row
    if (id == tail_id_p[0]) ti_s = tid;
  }

  // stage v[b] into LDS (coalesced float4)
  for (int p = tid; p < kS * (kH / 4); p += 256) {
    int t = p / (kH / 4), kk = (p % (kH / 4)) * 4;
    float4 val = *(const float4*)(v + ((size_t)b * kS + t) * kH + kk);
    float* dst = v_s + t * kVSTR + kk;
    dst[0] = val.x; dst[1] = val.y; dst[2] = val.z; dst[3] = val.w;
  }
  __syncthreads();
  const int hidx = hi_s, tidx = ti_s;

  // stage the two query rows
  for (int p = tid; p < 2 * (kH / 4); p += 256) {
    int r = p / (kH / 4), kk = (p % (kH / 4)) * 4;
    int srow = r ? tidx : hidx;
    float4 val = *(const float4*)(q + ((size_t)b * kS + srow) * kH + kk);
    float* dst = q_s + r * kH + kk;
    dst[0] = val.x; dst[1] = val.y; dst[2] = val.z; dst[3] = val.w;
  }
  __syncthreads();

  // 2*61 dot products + gate blend
  if (tid < 2 * kS) {
    int r = tid / kS, t = tid % kS;
    const float* qp = q_s + r * kH;       // broadcast within group
    const float* vp = v_s + t * kVSTR;
    float d0 = 0.0f;
    for (int k = 0; k < kH; ++k) d0 = fmaf(qp[k], vp[k], d0);
    float gt = g[b * kS + t];
    scr[r * 64 + t] = (d0 * (1.0f - gt) + gt * inv_s) * inv_scale;
  }
  __syncthreads();

  // softmax over t (61 values/row), serial per row: cheap + deterministic
  if (tid < 2) {
    float* row = scr + tid * 64;
    float m = row[0];
    for (int t = 1; t < kS; ++t) m = fmaxf(m, row[t]);
    float sum = 0.0f;
    for (int t = 0; t < kS; ++t) { float e = expf(row[t] - m); row[t] = e; sum += e; }
    float inv = 1.0f / sum;
    for (int t = 0; t < kS; ++t) row[t] *= inv;
  }
  __syncthreads();

  // weighted sums -> entity = [attn_out[head] | attn_out[tail]]
  for (int kk = tid; kk < kH; kk += 256) {
    float a0 = 0.0f, a1 = 0.0f;
    for (int t = 0; t < kS; ++t) {
      float vv = v_s[t * kVSTR + kk];
      a0 = fmaf(scr[t],      vv, a0);
      a1 = fmaf(scr[64 + t], vv, a1);
    }
    ent[kk]      = a0;
    ent[kH + kk] = a1;
  }
  __syncthreads();

  // logits: 23 x 1536 matvec, 8 partials per label
  if (tid < kNL * 8) {
    int l = tid >> 3, part = tid & 7;
    const float* wp = cls_w + (size_t)l * (2 * kH) + part * 192;
    const float* ep = ent + part * 192;
    float s = 0.0f;
    for (int k = 0; k < 192; ++k) s = fmaf(wp[k], ep[k], s);
    redc[l * 8 + part] = s;
  }
  __syncthreads();
  if (tid < kNL) {
    float s = cls_b[tid];
#pragma unroll
    for (int p2 = 0; p2 < 8; ++p2) s += redc[tid * 8 + p2];
    out[b * kNL + tid] = s;
  }
}

// ---------------------------------------------------------------------------
extern "C" void kernel_launch(void* const* d_in, const int* in_sizes, int n_in,
                              void* d_out, int out_size, void* d_ws, size_t ws_size,
                              hipStream_t stream) {
  const float* q      = (const float*)d_in[0];   // last_hidden_state
  const float* v      = (const float*)d_in[1];   // last_hidden_state_v
  const float* fc1_w  = (const float*)d_in[2];
  const float* fc1_b  = (const float*)d_in[3];
  const float* fc2_w  = (const float*)d_in[4];
  const float* fc2_b  = (const float*)d_in[5];
  const float* cls_w  = (const float*)d_in[6];
  const float* cls_b  = (const float*)d_in[7];
  const int* input_ids = (const int*)d_in[8];
  const int* head_id   = (const int*)d_in[9];
  const int* tail_id   = (const int*)d_in[10];
  float* out = (float*)d_out;

  // workspace: [fc1 swizzled bf16: 512*768*2 B][g: B*S floats]
  unsigned short* fc1sw = (unsigned short*)d_ws;
  float* g_ws = (float*)((char*)d_ws + (size_t)kDG * kH * sizeof(unsigned short));

  fc1_swizzle_kernel<<<(kDG * kH + 255) / 256, 256, 0, stream>>>(fc1_w, fc1sw);

  gate_kernel<<<(kB * kS) / 16, 256, 0, stream>>>(v, fc1sw, fc1_b, fc2_w, fc2_b, g_ws);

  size_t smem = (size_t)(kS * kVSTR + 128 + 2 * kH + 2 * kH + kNL * 8) * sizeof(float);
  attn_cls_kernel<<<kB, 256, smem, stream>>>(q, v, g_ws, input_ids,
                                             head_id, tail_id, cls_w, cls_b, out);
}
